// RTDETRTransformerPredictor_21655225106655
// MI455X (gfx1250) — matrix operand order repaired and checked by reading, not compiled, not run on hardware
//
#include <hip/hip_runtime.h>
#include <hip/hip_bf16.h>

// ---------------------------------------------------------------------------
// Multi-Scale Deformable Attention (RT-DETR) for gfx1250 (MI455X, wave32).
//   Pipeline:
//     1) v      = value @ Wv + bv            (bf16 WMMA GEMM, M=268800)
//     2) off    = query @ Woff + boff        (bf16 WMMA GEMM, M=9600, N=192)
//     3) logits = query @ Wattn + battn      (bf16 WMMA GEMM, M=9600, N=96)
//     4) mid    = softmax + bilinear deformable sampling (VALU + vmem)
//     5) out    = mid @ Wout + bout          (bf16 WMMA GEMM, M=9600)
// ---------------------------------------------------------------------------

#define BS    32
#define LQ    300
#define C_DIM 256
#define HEADS 8
#define NLVL  3
#define NPTS  4
#define DHEAD 32
#define LEN_V 8400

typedef __attribute__((ext_vector_type(16))) __bf16 v16bf;
typedef __attribute__((ext_vector_type(8)))  float  v8f;

// float -> bf16 native conversion (RNE) -> v_cvt_pk_bf16_f32 when paired.
__device__ __forceinline__ __bf16 f2bf(float f) { return static_cast<__bf16>(f); }

// ---------------------------------------------------------------------------
// GEMM: Co[M,N] = A[M,K] @ B[K,N] + bias[N]   (fp32 in/out, bf16 WMMA compute)
//   - 128 threads = 4 waves; wave w owns rows [blk*64 + w*16, +16)
//   - each wave accumulates the full N dimension: NT = N/16 tiles of v8f
//   - B converted to bf16 once per block, staged in LDS in fragment order
//   - inner loop: software-pipelined PAIRS of B fragments; loads for pair
//     (nt+2, nt+3) are issued before the WMMAs consuming (nt, nt+1), forcing
//     4 live fragment buffers so ds_load latency overlaps WMMA execution.
//   Requires: M % 64 == 0, K % 32 == 0, N == NT*16, NT even.
// ---------------------------------------------------------------------------
template <int NT, int KC>
__global__ void __launch_bounds__(128)
gemm_bias_wmma(const float* __restrict__ A, const float* __restrict__ B,
               const float* __restrict__ bias, float* __restrict__ Co) {
  constexpr int N      = NT * 16;
  constexpr int KSTEPS = KC / 32;
  extern __shared__ __align__(32) char lds_raw[];
  __bf16* ldsB = (__bf16*)lds_raw;

  const int tid = threadIdx.x;

  // ---- stage B into LDS in WMMA-fragment order -------------------------
  // fragment f = nt*KSTEPS + ks holds 512 bf16: [lane 0..31][elem 0..15]
  //   lanes 0-15 : n = nt*16 + lane      , k = ks*32 + (i&7) + 16*(i>>3)  (i = 0..15)
  //   lanes16-31 : n = nt*16 + (lane-16) , k = ks*32 + 8 + (i&7) + 16*(i>>3)
  constexpr int TOTAL = NT * KSTEPS * 512;
  for (int e = tid; e < TOTAL; e += 128) {
    int frag   = e >> 9;
    int within = e & 511;
    int lane   = within >> 4;
    int i      = within & 15;
    int ks     = frag % KSTEPS;
    int nt     = frag / KSTEPS;
    int k      = (ks << 5) + ((lane >> 4) << 3) + (i & 7) + ((i >> 3) << 4);
    int n      = (nt << 4) + (lane & 15);
    ldsB[e] = f2bf(B[(size_t)k * N + n]);
  }
  __syncthreads();

  const int wave  = tid >> 5;
  const int lane  = tid & 31;
  const int row0  = (blockIdx.x * 4 + wave) * 16;
  const int arow  = row0 + (lane & 15);
  const int khalf = (lane >> 4) ? 8 : 0;   // K sub-block per lane half

  v8f acc[NT];
#pragma unroll
  for (int t = 0; t < NT; ++t) acc[t] = (v8f){0.f, 0.f, 0.f, 0.f, 0.f, 0.f, 0.f, 0.f};

  const float* arow_base = A + (size_t)arow * KC + khalf;
  for (int ks = 0; ks < KSTEPS; ++ks) {
    const float* ap = arow_base + (ks << 5);
    // prefetch next K-step of A into cache (global_prefetch_b8)
    __builtin_prefetch(ap + 32, 0, 0);

    // A fragment (two 16B-aligned float4 pairs -> global_load_b128):
    //   elems 0..7 -> K {khalf..khalf+7}, elems 8..15 -> +16
    const float4* ap4 = (const float4*)ap;
    const float4 a0 = ap4[0], a1 = ap4[1];   // khalf + 0..7
    const float4 a2 = ap4[4], a3 = ap4[5];   // khalf + 16..23
    v16bf afrag;
    afrag[0]  = f2bf(a0.x); afrag[1]  = f2bf(a0.y);
    afrag[2]  = f2bf(a0.z); afrag[3]  = f2bf(a0.w);
    afrag[4]  = f2bf(a1.x); afrag[5]  = f2bf(a1.y);
    afrag[6]  = f2bf(a1.z); afrag[7]  = f2bf(a1.w);
    afrag[8]  = f2bf(a2.x); afrag[9]  = f2bf(a2.y);
    afrag[10] = f2bf(a2.z); afrag[11] = f2bf(a2.w);
    afrag[12] = f2bf(a3.x); afrag[13] = f2bf(a3.y);
    afrag[14] = f2bf(a3.z); afrag[15] = f2bf(a3.w);

    // B-fragment pointer for this K-step; tile stride is compile-time.
    const __bf16*  bp  = ldsB + (ks << 9) + (lane << 4);
    constexpr int  ntS = KSTEPS << 9;

    // Pipelined pairs: loads for (nt+2, nt+3) issue before WMMAs of (nt, nt+1)
    v16bf b0 = *(const v16bf*)(bp);
    v16bf b1 = *(const v16bf*)(bp + ntS);
#pragma unroll
    for (int nt = 0; nt < NT; nt += 2) {
      v16bf n0 = b0, n1 = b1;
      if (nt + 2 < NT) {
        n0 = *(const v16bf*)(bp + (nt + 2) * ntS);
        n1 = *(const v16bf*)(bp + (nt + 3) * ntS);
      }
      acc[nt] = __builtin_amdgcn_wmma_f32_16x16x32_bf16(
          false, afrag, false, b0, (short)0, acc[nt], false, false);
      acc[nt + 1] = __builtin_amdgcn_wmma_f32_16x16x32_bf16(
          false, afrag, false, b1, (short)0, acc[nt + 1], false, false);
      b0 = n0; b1 = n1;
    }
  }

  // ---- store: VGPR r -> row0 + r (+8 for lanes 16-31), col = lane&15 ---
  const int rbase = row0 + ((lane >> 4) << 3);
  const int cl    = lane & 15;
#pragma unroll
  for (int nt = 0; nt < NT; ++nt) {
    const int n  = (nt << 4) + cl;
    const float bv = bias[n];
#pragma unroll
    for (int r = 0; r < 8; ++r) {
      Co[(size_t)(rbase + r) * N + n] = acc[nt][r] + bv;
    }
  }
}

// ---------------------------------------------------------------------------
// Deformable sampling: one block per (b,q); wave h handles head h; lane = d.
//   v      : [BS, LEN_V, 256]   (head h occupies cols h*32..h*32+31)
//   offs   : [BS*LQ, 192]       (h, l, p, {x,y})
//   logits : [BS*LQ, 96]        (h, l*P+p) -> softmax over 12
//   refp   : [BS*LQ, L, 2]
//   out    : [BS*LQ, 256]
// ---------------------------------------------------------------------------
__global__ void __launch_bounds__(256)
msda_sample(const float* __restrict__ v, const float* __restrict__ offs,
            const float* __restrict__ logits, const float* __restrict__ refp,
            float* __restrict__ out) {
  const int bq = blockIdx.x;
  const int b  = bq / LQ;
  const int h  = threadIdx.x >> 5;
  const int d  = threadIdx.x & 31;

  const int dims[NLVL]   = {20, 40, 80};
  const int starts[NLVL] = {0, 400, 2000};

  // softmax over the 12 (level, point) logits of this head (broadcast loads)
  const float* lg = logits + (size_t)bq * (HEADS * NLVL * NPTS) + h * (NLVL * NPTS);
  float w[NLVL * NPTS];
  float mx = -3.0e38f;
#pragma unroll
  for (int i = 0; i < NLVL * NPTS; ++i) { w[i] = lg[i]; mx = fmaxf(mx, w[i]); }
  float s = 0.f;
#pragma unroll
  for (int i = 0; i < NLVL * NPTS; ++i) { w[i] = __expf(w[i] - mx); s += w[i]; }
  const float invs = 1.f / s;

  const float* vb = v + ((size_t)b * LEN_V) * C_DIM + h * DHEAD + d;
  const float* op = offs + (size_t)bq * (HEADS * NLVL * NPTS * 2) + h * (NLVL * NPTS * 2);

  float acc = 0.f;
#pragma unroll
  for (int l = 0; l < NLVL; ++l) {
    const int   Wl = dims[l], Hl = dims[l];
    const float rx = refp[(size_t)(bq * NLVL + l) * 2 + 0];
    const float ry = refp[(size_t)(bq * NLVL + l) * 2 + 1];
    const float* vl = vb + (size_t)starts[l] * C_DIM;
#pragma unroll
    for (int p = 0; p < NPTS; ++p) {
      const float ox = op[(l * NPTS + p) * 2 + 0];
      const float oy = op[(l * NPTS + p) * 2 + 1];
      // (ref + off/norm) * dim - 0.5  ==  ref*dim + off - 0.5
      const float x  = rx * (float)Wl + ox - 0.5f;
      const float y  = ry * (float)Hl + oy - 0.5f;
      const float xf = floorf(x), yf = floorf(y);
      const float fx = x - xf,    fy = y - yf;
      const int x0 = (int)xf, y0 = (int)yf;
      const int x1 = x0 + 1,  y1 = y0 + 1;
      const bool vx0 = (x0 >= 0) & (x0 < Wl);
      const bool vx1 = (x1 >= 0) & (x1 < Wl);
      const bool vy0 = (y0 >= 0) & (y0 < Hl);
      const bool vy1 = (y1 >= 0) & (y1 < Hl);
      float g00 = 0.f, g01 = 0.f, g10 = 0.f, g11 = 0.f;
      if (vy0 & vx0) g00 = vl[(size_t)(y0 * Wl + x0) * C_DIM];
      if (vy0 & vx1) g01 = vl[(size_t)(y0 * Wl + x1) * C_DIM];
      if (vy1 & vx0) g10 = vl[(size_t)(y1 * Wl + x0) * C_DIM];
      if (vy1 & vx1) g11 = vl[(size_t)(y1 * Wl + x1) * C_DIM];
      const float bil = (g00 * (1.f - fx) + g01 * fx) * (1.f - fy) +
                        (g10 * (1.f - fx) + g11 * fx) * fy;
      acc += w[l * NPTS + p] * bil;
    }
  }
  out[(size_t)bq * C_DIM + h * DHEAD + d] = acc * invs;
}

// ---------------------------------------------------------------------------
extern "C" void kernel_launch(void* const* d_in, const int* in_sizes, int n_in,
                              void* d_out, int out_size, void* d_ws, size_t ws_size,
                              hipStream_t stream) {
  (void)in_sizes; (void)n_in; (void)out_size; (void)ws_size;

  const float* query = (const float*)d_in[0];   // [32,300,256]
  const float* refp  = (const float*)d_in[1];   // [32,300,3,2]
  const float* value = (const float*)d_in[2];   // [32,8400,256]
  const float* Wv    = (const float*)d_in[3];   // [256,256]
  const float* bv    = (const float*)d_in[4];   // [256]
  const float* Woff  = (const float*)d_in[5];   // [256,192]
  const float* boff  = (const float*)d_in[6];   // [192]
  const float* Wattn = (const float*)d_in[7];   // [256,96]
  const float* battn = (const float*)d_in[8];   // [96]
  const float* Wout  = (const float*)d_in[9];   // [256,256]
  const float* bout  = (const float*)d_in[10];  // [256]
  float* out = (float*)d_out;                   // [32,300,256]

  // workspace layout (floats)
  float* ws       = (float*)d_ws;
  float* v_ws     = ws;                                  // 32*8400*256 = 68,812,800
  float* off_ws   = v_ws   + (size_t)BS * LEN_V * C_DIM; //  9600*192   =  1,843,200
  float* logit_ws = off_ws + (size_t)BS * LQ * 192;      //  9600*96    =    921,600
  float* mid_ws   = logit_ws + (size_t)BS * LQ * 96;     //  9600*256   =  2,457,600

  const int Mv = BS * LEN_V;   // 268800  (divisible by 64)
  const int Mq = BS * LQ;      // 9600    (divisible by 64)

  // 1) value projection (dominant GEMM, HBM-roofline bound with bf16 WMMA)
  gemm_bias_wmma<16, 256><<<dim3(Mv / 64), dim3(128), 256 * 256 * 2, stream>>>(
      value, Wv, bv, v_ws);

  // 2) sampling offsets
  gemm_bias_wmma<12, 256><<<dim3(Mq / 64), dim3(128), 256 * 192 * 2, stream>>>(
      query, Woff, boff, off_ws);

  // 3) attention logits
  gemm_bias_wmma<6, 256><<<dim3(Mq / 64), dim3(128), 256 * 96 * 2, stream>>>(
      query, Wattn, battn, logit_ws);

  // 4) softmax + bilinear deformable sampling
  msda_sample<<<dim3(Mq), dim3(256), 0, stream>>>(
      v_ws, off_ws, logit_ws, refp, mid_ws);

  // 5) output projection
  gemm_bias_wmma<16, 256><<<dim3(Mq / 64), dim3(128), 256 * 256 * 2, stream>>>(
      mid_ws, Wout, bout, out);
}
